// FILIPInfoNCE_77068893160126
// MI455X (gfx1250) — matrix operand order repaired and checked by reading, not compiled, not run on hardware
//
#include <hip/hip_runtime.h>

// ---------------------------------------------------------------------------
// FILIP token-wise InfoNCE on MI455X (gfx1250, wave32, WMMA + TDM).
//
// sim = norm(t1)(12544x256) @ norm(t2)^T, as 64x64 blocks of 196x196.
// 80.6 GFLOP vs 25.7 MB input -> compute bound; bf16 WMMA (K=32/issue).
//
// Data movement:
//  * B panel (100 KB, one b2) staged into LDS once per workgroup by the
//    Tensor Data Mover (tensor_load_to_lds + s_wait_tensorcnt), with TDM
//    row padding (512B row + 16B pad = 528B stride) for conflict-free reads.
//  * Each wave owns TWO 16-row A strips in registers, so every B fragment
//    read from LDS feeds two back-to-back v_wmma ops: 512B LDS per WMMA
//    instead of 1KB, and half the per-WMMA dscnt waits.
// ---------------------------------------------------------------------------

typedef __attribute__((ext_vector_type(16))) __bf16 v16bf;
typedef __attribute__((ext_vector_type(8)))  __bf16 v8bf;
typedef __attribute__((ext_vector_type(8)))  float  v8f;
typedef __attribute__((ext_vector_type(4)))  unsigned int v4u;
typedef __attribute__((ext_vector_type(8)))  int    v8i_t;
typedef __attribute__((ext_vector_type(4)))  int    v4i_t;

#define BATCH 64
#define PTOK  196
#define DIM   256
#define NSTRIP 14           // 7 waves x 2 strips (strip 13 = padding)
#define NTILE 13            // valid column tiles
#define COLP  208
#define NWAVE 7
#define NTHREADS (NWAVE * 32)
#define BROWW 132           // padded B-panel row stride in DWORDs (528 B)

// Dynamic LDS partition (32-bit words)
#define W_BPANEL  0
#define W_TILE    (W_BPANEL + PTOK * BROWW)       // 14 strips * 256 f32
#define W_CMAX    (W_TILE   + NSTRIP * 256)
#define W_CIDX    (W_CMAX   + NSTRIP * COLP)
#define W_RARR    (W_CIDX   + NSTRIP * COLP)
#define W_RIDX    (W_RARR   + COLP)
#define W_TOTAL   (W_RIDX   + COLP)
#define SMEM_BYTES (W_TOTAL * 4)

#define USE_TDM (__has_builtin(__builtin_amdgcn_tensor_load_to_lds) && \
                 __has_builtin(__builtin_amdgcn_s_wait_tensorcnt))

// ---------------------------------------------------------------------------
// Kernel 1: L2-normalize rows (f32) and convert to bf16.
// ---------------------------------------------------------------------------
__global__ __launch_bounds__(256)
void normalize_to_bf16(const float* __restrict__ t1,
                       const float* __restrict__ t2,
                       __bf16* __restrict__ nt1,
                       __bf16* __restrict__ nt2) {
    const int row = blockIdx.x;
    const int tid = threadIdx.x;
    const int nrows = BATCH * PTOK;

    const float* src = (row < nrows) ? (t1 + (size_t)row * DIM)
                                     : (t2 + (size_t)(row - nrows) * DIM);
    __bf16* dst = (row < nrows) ? (nt1 + (size_t)row * DIM)
                                : (nt2 + (size_t)(row - nrows) * DIM);

    float x = src[tid];
    __shared__ float red[256];
    red[tid] = x * x;
    __syncthreads();
    #pragma unroll
    for (int s = 128; s > 0; s >>= 1) {
        if (tid < s) red[tid] += red[tid + s];
        __syncthreads();
    }
    const float nrm   = sqrtf(red[0]);
    const float scale = 1.0f / fmaxf(nrm, 1e-12f);
    dst[tid] = (__bf16)(x * scale);
}

// ---------------------------------------------------------------------------
// Kernel 2: per-(b1,b2) 196x196 block. 7 waves; wave w owns row strips
// 2w and 2w+1 (two A-fragment sets + two accumulators -> 2 WMMA per B frag).
// ---------------------------------------------------------------------------
__global__ __launch_bounds__(NTHREADS)
void filip_sim(const __bf16* __restrict__ nt1,
               const __bf16* __restrict__ nt2,
               float* __restrict__ simMat,     // [64,64]
               float* __restrict__ outIdx12,   // [64,196] (diag only)
               float* __restrict__ outIdx21) { // [64,196] (diag only)
    extern __shared__ __align__(16) unsigned int smemW[];
    __bf16* bpanel    = (__bf16*)(smemW + W_BPANEL);  // rows of 264 bf16
    float*  tileBuf   = (float*) (smemW + W_TILE);
    float*  colMaxP   = (float*) (smemW + W_CMAX);
    int*    colIdxP   = (int*)   (smemW + W_CIDX);
    float*  rowArr    = (float*) (smemW + W_RARR);
    int*    rowIdxArr = (int*)   (smemW + W_RIDX);

    const int b1   = blockIdx.x >> 6;
    const int b2   = blockIdx.x & 63;
    const int tid  = threadIdx.x;
    const int w    = tid >> 5;      // wave id 0..6
    const int lane = tid & 31;
    const int half = lane >> 4;     // 0: lanes 0-15, 1: lanes 16-31
    const int l16  = lane & 15;

    const int s0 = 2 * w;           // strips owned by this wave
    const int s1 = 2 * w + 1;       // s1 == 13 is padding (fully masked)
    const int i0a = s0 * 16;
    const int i0b = s1 * 16;

    const __bf16* Abase = nt1 + (size_t)b1 * (PTOK * DIM);
    const __bf16* Bbase = nt2 + (size_t)b2 * (PTOK * DIM);

#if USE_TDM
    // --- Stage the whole B panel (196x256 bf16) LDS-padded via the TDM. ---
    if (w == 0) {
        const unsigned long long ga = (unsigned long long)(size_t)Bbase;
        const unsigned int ldsOff   = (unsigned int)(size_t)(void*)bpanel;
        v4u g0;
        g0[0] = 1u;                                   // count=1 (valid user D#)
        g0[1] = ldsOff;                               // lds_addr
        g0[2] = (unsigned int)ga;                     // global_addr[31:0]
        g0[3] = ((unsigned int)(ga >> 32) & 0x01FFFFFFu) | (2u << 30); // type=2
        v8i_t g1;
        g1[0] = (int)((1u << 16)      // data_size = 1 -> 2-byte elements
                    | (1u << 20)      // pad_enable
                    | (6u << 22)      // pad_interval: 128 DWORDs (512B row)
                    | (3u << 25));    // pad_amount: 4 DWORDs (16 B)
        g1[1] = (int)(DIM  << 16);    // tensor_dim0 lo16 (=256)
        g1[2] = (int)(PTOK << 16);    // tensor_dim0 hi | tensor_dim1 lo (=196)
        g1[3] = (int)(DIM  << 16);    // tensor_dim1 hi | tile_dim0 (=256)
        g1[4] = (int)PTOK;            // tile_dim1 (=196), tile_dim2=0
        g1[5] = (int)DIM;             // tensor_dim0_stride lo (=256)
        g1[6] = 0;
        g1[7] = 0;
        v4i_t g2 = {};
        v4i_t g3 = {};
#if __clang_major__ >= 23
        v8i_t g4 = {};
        __builtin_amdgcn_tensor_load_to_lds(g0, g1, g2, g3, g4, 0);
#else
        __builtin_amdgcn_tensor_load_to_lds(g0, g1, g2, g3, 0);
#endif
    }
#else
    // Fallback: cooperative copy with manual 528B row stride.
    {
        const unsigned int* gsrc = (const unsigned int*)Bbase;
        unsigned int* ldst = smemW + W_BPANEL;
        for (int i = tid; i < PTOK * 128; i += NTHREADS) {
            const int r = i >> 7, c = i & 127;
            ldst[r * BROWW + c] = gsrc[r * 128 + c];
        }
    }
#endif

    // Preload A fragments for both strips (16x32 bf16, ISA VGPR layout:
    // lanes 0-15 hold K {0..7,16..23}; lanes 16-31 hold K {8..15,24..31}).
    int arow0 = i0a + l16; if (arow0 > PTOK - 1) arow0 = PTOK - 1;
    int arow1 = i0b + l16; if (arow1 > PTOK - 1) arow1 = PTOK - 1;
    v16bf afr0[8], afr1[8];
    #pragma unroll
    for (int kc = 0; kc < 8; ++kc) {
        const __bf16* rp0 = Abase + (size_t)arow0 * DIM + kc * 32 + half * 8;
        const __bf16* rp1 = Abase + (size_t)arow1 * DIM + kc * 32 + half * 8;
        v8bf lo0 = *reinterpret_cast<const v8bf*>(rp0);
        v8bf hi0 = *reinterpret_cast<const v8bf*>(rp0 + 16);
        v8bf lo1 = *reinterpret_cast<const v8bf*>(rp1);
        v8bf hi1 = *reinterpret_cast<const v8bf*>(rp1 + 16);
        #pragma unroll
        for (int e = 0; e < 8; ++e) {
            afr0[kc][e] = lo0[e]; afr0[kc][e + 8] = hi0[e];
            afr1[kc][e] = lo1[e]; afr1[kc][e + 8] = hi1[e];
        }
    }

#if USE_TDM
    if (w == 0) __builtin_amdgcn_s_wait_tensorcnt((unsigned short)0);
#endif
    __syncthreads();   // B panel visible to all waves

    float rowMax0 = -3.4e38f, rowMax1 = -3.4e38f;
    int   rowIdx0 = 0,        rowIdx1 = 0;
    float* myTile0 = tileBuf + s0 * 256;
    float* myTile1 = tileBuf + s1 * 256;

    for (int tj = 0; tj < NTILE; ++tj) {
        const int j0 = tj * 16;
        int brow = j0 + l16;
        if (brow > PTOK - 1) brow = PTOK - 1;

        // K loop: each B fragment (2 x ds_load_b128) feeds 2 WMMAs.
        const __bf16* bp = bpanel + (size_t)brow * (2 * BROWW);
        v8f c0 = {}, c1 = {};
        #pragma unroll
        for (int kc = 0; kc < 8; ++kc) {
            const __bf16* rp = bp + kc * 32 + half * 8;
            v8bf lo = *reinterpret_cast<const v8bf*>(rp);
            v8bf hi = *reinterpret_cast<const v8bf*>(rp + 16);
            v16bf bfr;
            #pragma unroll
            for (int e = 0; e < 8; ++e) { bfr[e] = lo[e]; bfr[e + 8] = hi[e]; }
            c0 = __builtin_amdgcn_wmma_f32_16x16x32_bf16(
                     false, afr0[kc], false, bfr, (short)0, c0, false, false);
            c1 = __builtin_amdgcn_wmma_f32_16x16x32_bf16(
                     false, afr1[kc], false, bfr, (short)0, c1, false, false);
        }

        // C/D layout: VGPR r holds (m = r + half*8, n = l16).
        #pragma unroll
        for (int r = 0; r < 8; ++r) {
            myTile0[(r + half * 8) * 16 + l16] = c0[r];
            myTile1[(r + half * 8) * 16 + l16] = c1[r];
        }
        asm volatile("s_wait_dscnt 0x0" ::: "memory");   // same-wave LDS RAW

        if (half == 0) {
            // Row scans for both strips: m = l16.
            #pragma unroll
            for (int n = 0; n < 16; ++n) {
                if (j0 + n < PTOK) {                     // wave-uniform
                    float v0 = myTile0[l16 * 16 + n];
                    if (v0 > rowMax0) { rowMax0 = v0; rowIdx0 = j0 + n; }
                    float v1 = myTile1[l16 * 16 + n];
                    if (v1 > rowMax1) { rowMax1 = v1; rowIdx1 = j0 + n; }
                }
            }
        } else {
            // Column scans: n = l16; distinct columns each tj -> partials.
            float cm0 = -3.4e38f; int ci0 = 0;
            float cm1 = -3.4e38f; int ci1 = 0;
            #pragma unroll
            for (int m = 0; m < 16; ++m) {
                if (i0a + m < PTOK) {                    // wave-uniform
                    float v = myTile0[m * 16 + l16];
                    if (v > cm0) { cm0 = v; ci0 = i0a + m; }
                }
                if (i0b + m < PTOK) {                    // wave-uniform
                    float v = myTile1[m * 16 + l16];
                    if (v > cm1) { cm1 = v; ci1 = i0b + m; }
                }
            }
            colMaxP[s0 * COLP + j0 + l16] = cm0;
            colIdxP[s0 * COLP + j0 + l16] = ci0;
            colMaxP[s1 * COLP + j0 + l16] = cm1;
            colIdxP[s1 * COLP + j0 + l16] = ci1;
        }
    }

    if (half == 0) {
        if (i0a + l16 < PTOK) {
            rowArr[i0a + l16]    = rowMax0;
            rowIdxArr[i0a + l16] = rowIdx0;
        }
        if (i0b + l16 < PTOK) {
            rowArr[i0b + l16]    = rowMax1;
            rowIdxArr[i0b + l16] = rowIdx1;
        }
    }
    __syncthreads();

    // Deterministic serial sum of 196 row maxes -> mean.
    if (tid == 0) {
        float s = 0.0f;
        for (int r = 0; r < PTOK; ++r) s += rowArr[r];
        simMat[b1 * BATCH + b2] = s * (1.0f / PTOK);
    }

    // Diagonal blocks: emit argmax indices (first-occurrence kept by strict
    // '>' with ascending scan order over n, m, and strips).
    if (b1 == b2 && tid < PTOK) {
        outIdx12[b1 * PTOK + tid] = (float)rowIdxArr[tid];
        float cm = -3.4e38f; int ci = 0;
        for (int wv = 0; wv < NSTRIP; ++wv) {
            float v = colMaxP[wv * COLP + tid];
            if (v > cm) { cm = v; ci = colIdxP[wv * COLP + tid]; }
        }
        outIdx21[b1 * PTOK + tid] = (float)ci;
    }
}

// ---------------------------------------------------------------------------
// Kernel 3: InfoNCE over the 64x64 logits (1 block, 64 threads).
// ---------------------------------------------------------------------------
__global__ __launch_bounds__(64)
void filip_loss(const float* __restrict__ simMat, float* __restrict__ outLoss) {
    __shared__ float contrib[BATCH];
    const int i = threadIdx.x;
    const float invT = 10.0f; // 1 / 0.1

    float m = -3.4e38f;
    for (int j = 0; j < BATCH; ++j)
        m = fmaxf(m, simMat[i * BATCH + j] * invT);
    float s = 0.0f;
    for (int j = 0; j < BATCH; ++j)
        s += __expf(simMat[i * BATCH + j] * invT - m);
    const float lse = m + __logf(s);
    contrib[i] = lse - simMat[i * BATCH + i] * invT;
    __syncthreads();
    if (i == 0) {
        float acc = 0.0f;
        for (int j = 0; j < BATCH; ++j) acc += contrib[j];
        outLoss[0] = acc * (1.0f / BATCH);
    }
}

// ---------------------------------------------------------------------------
// Host-side launcher.
// d_out layout (25089 floats): [0] loss, [1..12544] idx1_2, [12545..] idx2_1.
// ---------------------------------------------------------------------------
extern "C" void kernel_launch(void* const* d_in, const int* in_sizes, int n_in,
                              void* d_out, int out_size, void* d_ws, size_t ws_size,
                              hipStream_t stream) {
    const float* t1 = (const float*)d_in[0];   // [64,196,256] f32
    const float* t2 = (const float*)d_in[1];   // [64,196,256] f32
    // d_in[2] (labels) and d_in[3] (epoch) are unused by the reference math.

    const size_t nElem = (size_t)BATCH * PTOK * DIM;            // 3,211,264
    char*   ws     = (char*)d_ws;
    __bf16* nt1    = (__bf16*)ws;                               // 6.42 MB
    __bf16* nt2    = (__bf16*)(ws + nElem * sizeof(__bf16));
    float*  simMat = (float*)(ws + 2 * nElem * sizeof(__bf16)); // 16 KB

    float* out      = (float*)d_out;
    float* outLoss  = out;
    float* outIdx12 = out + 1;
    float* outIdx21 = out + 1 + BATCH * PTOK;

    normalize_to_bf16<<<2 * BATCH * PTOK, 256, 0, stream>>>(t1, t2, nt1, nt2);
    filip_sim<<<BATCH * BATCH, NTHREADS, SMEM_BYTES, stream>>>(
        nt1, nt2, simMat, outIdx12, outIdx21);
    filip_loss<<<1, BATCH, 0, stream>>>(simMat, outLoss);
}
